// MyLSTM_4475355922901
// MI455X (gfx1250) — compile-verified, matrix operand order checked by reading, not run on hardware
//
#include <hip/hip_runtime.h>
#include <hip/hip_bf16.h>
#include <stdint.h>

#define B   64
#define T   512
#define IN  128
#define H   1024
#define G4H 4096
#define NWG_REC 32

#define KCH  128              // K-chunk staged per LDS buffer
#define APAD 8                // row padding (bf16 elems) to de-conflict LDS banks
#define ALD  (KCH + APAD)     // staged row stride = 136 elems = 272 B

typedef __attribute__((ext_vector_type(16))) __bf16 v16bf;
typedef __attribute__((ext_vector_type(8)))  float  v8f;
typedef __attribute__((ext_vector_type(4)))  int    v4i;

union Frag { v16bf v; uint4 q[2]; };

#if __has_builtin(__builtin_amdgcn_global_load_async_to_lds_b128) && \
    __has_builtin(__builtin_amdgcn_s_wait_asynccnt)
#define USE_ASYNC_LDS 1
#else
#define USE_ASYNC_LDS 0
#endif

__device__ inline void async_cp_b128(const unsigned short* g, unsigned short* l) {
#if USE_ASYNC_LDS
  __builtin_amdgcn_global_load_async_to_lds_b128(
      (__attribute__((address_space(1))) v4i*)(uintptr_t)g,
      (__attribute__((address_space(3))) v4i*)(uintptr_t)l, 0, 0);
#else
  *(uint4*)l = *(const uint4*)g;
#endif
}
__device__ inline void async_wait_all() {
#if USE_ASYNC_LDS
  __builtin_amdgcn_s_wait_asynccnt(0);
#endif
}

// stage a 64-row x 128-col bf16 chunk: src row stride = lda, dst row stride = ALD
__device__ inline void stage_chunk(const unsigned short* __restrict__ src, int lda,
                                   unsigned short* __restrict__ dst, int tid) {
  #pragma unroll
  for (int i = 0; i < 4; ++i) {
    int idx = tid + i * 256;            // 0..1023 : 64 rows x 16 x b128
    int row = idx >> 4, seg = idx & 15;
    async_cp_b128(src + (size_t)row * lda + seg * 8, dst + row * ALD + seg * 8);
  }
}

__device__ inline unsigned short f2bf(float f) {
  union { float f; unsigned int u; } x; x.f = f;
  unsigned int r = x.u + 0x7fffu + ((x.u >> 16) & 1u);
  return (unsigned short)(r >> 16);
}
__device__ inline float bf2f(unsigned short b) {
  union { unsigned int u; float f; } x; x.u = ((unsigned int)b) << 16;
  return x.f;
}
__device__ inline float sigf(float x) { return 1.0f / (1.0f + __expf(-x)); }

// ---------------------------------------------------------------- conversions
__global__ void cvt_f32_bf16(const float* __restrict__ s, unsigned short* __restrict__ d, int n) {
  int i = blockIdx.x * blockDim.x + threadIdx.x;
  if (i < n) d[i] = f2bf(s[i]);
}

// -------------------------------------------------- big input-projection GEMM
// xp[M=B*T, 4H] = A_bf16[M,K] @ W_bf16[4H,K]^T + b_ih + b_hh   (f32 accumulate)
// A block (64 rows x K) is async-staged into LDS in 128-col chunks (double buffer).
__global__ void __launch_bounds__(256) gemm_xp(
    const unsigned short* __restrict__ A, int K,
    const unsigned short* __restrict__ W,
    const float* __restrict__ bih, const float* __restrict__ bhh,
    float* __restrict__ xp) {
  __shared__ __align__(16) unsigned short astage[2 * 64 * ALD];
  const int tid    = threadIdx.x;
  const int lane   = tid & 31;
  const int wave   = tid >> 5;
  const int col    = blockIdx.x * 128 + wave * 16 + (lane & 15);
  const int m0     = blockIdx.y * 64;
  const int koff   = (lane >> 4) * 8;
  const int rowoff = (lane >> 4) << 3;

  float bias = bih[col] + bhh[col];
  v8f acc[4];
  #pragma unroll
  for (int mt = 0; mt < 4; ++mt)
    #pragma unroll
    for (int r = 0; r < 8; ++r) acc[mt][r] = bias;

  const unsigned short* abase = A + (size_t)m0 * K;
  const unsigned short* bp    = W + (size_t)col * K + koff;
  const int nch = K / KCH;

  stage_chunk(abase, K, astage, tid);
  async_wait_all();
  __syncthreads();

  for (int kc = 0; kc < nch; ++kc) {
    if (kc + 1 < nch)
      stage_chunk(abase + (kc + 1) * KCH, K, astage + ((kc + 1) & 1) * 64 * ALD, tid);
    const unsigned short* sb = astage + (kc & 1) * 64 * ALD;
    #pragma unroll
    for (int kk = 0; kk < KCH; kk += 32) {
      const int k0 = kc * KCH + kk;
      Frag bf;
      bf.q[0] = *(const uint4*)(bp + k0);
      bf.q[1] = *(const uint4*)(bp + k0 + 16);
      __builtin_prefetch(bp + k0 + 32, 0, 1);
      #pragma unroll
      for (int mt = 0; mt < 4; ++mt) {
        const unsigned short* ap = sb + (mt * 16 + (lane & 15)) * ALD + kk + koff;
        Frag af;
        af.q[0] = *(const uint4*)ap;
        af.q[1] = *(const uint4*)(ap + 16);
        acc[mt] = __builtin_amdgcn_wmma_f32_16x16x32_bf16(
            false, af.v, false, bf.v, (short)0, acc[mt], false, false);
      }
    }
    async_wait_all();
    __syncthreads();
  }

  #pragma unroll
  for (int mt = 0; mt < 4; ++mt)
    #pragma unroll
    for (int r = 0; r < 8; ++r)
      xp[(size_t)(m0 + mt * 16 + r + rowoff) * G4H + col] = acc[mt][r];
}

// ----------------------------------------------------------------- state init
__global__ void init_state(unsigned short* hA, unsigned short* hB, float* c, unsigned int* bar) {
  int i = blockIdx.x * blockDim.x + threadIdx.x;
  if (i < B * H) { hA[i] = 0; hB[i] = 0; c[i] = 0.0f; }
  if (i < 2) bar[i] = 0u;
}

// --------------------------------------------------- persistent recurrent kernel
// 32 WGs, each owns H-columns [n0, n0+32). Per step: gates = xp[t] + h @ Whh^T.
// h (64x1024 bf16) is async-staged into LDS in 128-col double-buffered chunks.
__global__ void __launch_bounds__(256) lstm_rec(
    const float* __restrict__ xp, const unsigned short* __restrict__ Whh,
    unsigned short* hA, unsigned short* hB, float* __restrict__ c,
    unsigned short* __restrict__ seq, unsigned int* bar) {
  __shared__ __align__(16) unsigned short hstage[2 * 64 * ALD];
  __shared__ float gbuf[64 * 128];                 // [B][4 gates x 32 cols]
  const int tid    = threadIdx.x;
  const int lane   = tid & 31;
  const int wave   = tid >> 5;
  const int n0     = blockIdx.x * 32;
  const int g      = wave >> 1;
  const int j0     = (wave & 1) * 16;
  const int gcol   = g * H + n0 + j0 + (lane & 15);   // gate column / Whh row
  const int lcol   = g * 32 + j0 + (lane & 15);       // LDS column in gbuf
  const int koff   = (lane >> 4) * 8;
  const int rowoff = (lane >> 4) << 3;
  const unsigned short* wp = Whh + (size_t)gcol * H + koff;

  for (int t = 0; t < T; ++t) {
    const unsigned short* hp = (t & 1) ? hB : hA;
    unsigned short*       hn = (t & 1) ? hA : hB;

    v8f acc[4];
    #pragma unroll
    for (int mt = 0; mt < 4; ++mt)
      #pragma unroll
      for (int r = 0; r < 8; ++r)
        acc[mt][r] = xp[((size_t)(mt * 16 + r + rowoff) * T + t) * G4H + gcol];

    stage_chunk(hp, H, hstage, tid);
    async_wait_all();
    __syncthreads();

    for (int kc = 0; kc < H / KCH; ++kc) {
      if (kc + 1 < H / KCH)
        stage_chunk(hp + (kc + 1) * KCH, H, hstage + ((kc + 1) & 1) * 64 * ALD, tid);
      const unsigned short* sb = hstage + (kc & 1) * 64 * ALD;
      #pragma unroll
      for (int kk = 0; kk < KCH; kk += 32) {
        const int k0 = kc * KCH + kk;
        Frag bf;
        bf.q[0] = *(const uint4*)(wp + k0);
        bf.q[1] = *(const uint4*)(wp + k0 + 16);
        __builtin_prefetch(wp + k0 + 32, 0, 1);
        #pragma unroll
        for (int mt = 0; mt < 4; ++mt) {
          const unsigned short* ap = sb + (mt * 16 + (lane & 15)) * ALD + kk + koff;
          Frag af;
          af.q[0] = *(const uint4*)ap;
          af.q[1] = *(const uint4*)(ap + 16);
          acc[mt] = __builtin_amdgcn_wmma_f32_16x16x32_bf16(
              false, af.v, false, bf.v, (short)0, acc[mt], false, false);
        }
      }
      async_wait_all();
      __syncthreads();
    }

    #pragma unroll
    for (int mt = 0; mt < 4; ++mt)
      #pragma unroll
      for (int r = 0; r < 8; ++r)
        gbuf[(mt * 16 + r + rowoff) * 128 + lcol] = acc[mt][r];
    __syncthreads();

    // elementwise cell: 64x32 = 2048 (b, j) pairs / 256 threads
    #pragma unroll
    for (int p8 = 0; p8 < 8; ++p8) {
      int p = tid + p8 * 256;
      int b = p >> 5, j = p & 31;
      float i_ = gbuf[b * 128 +       j];
      float f_ = gbuf[b * 128 +  32 + j];
      float g_ = gbuf[b * 128 +  64 + j];
      float o_ = gbuf[b * 128 +  96 + j];
      int hidx = b * H + n0 + j;
      float cc = sigf(f_) * c[hidx] + sigf(i_) * tanhf(g_);
      float hh = sigf(o_) * tanhf(cc);
      c[hidx] = cc;
      unsigned short hb = f2bf(hh);
      hn[hidx] = hb;
      seq[((size_t)b * T + t) * H + n0 + j] = hb;
    }

    // device-wide barrier before next timestep
    __threadfence();
    __syncthreads();
    if (tid == 0) {
      unsigned int gen  = __hip_atomic_load(&bar[1], __ATOMIC_ACQUIRE, __HIP_MEMORY_SCOPE_AGENT);
      unsigned int prev = __hip_atomic_fetch_add(&bar[0], 1u, __ATOMIC_ACQ_REL, __HIP_MEMORY_SCOPE_AGENT);
      if (prev == NWG_REC - 1) {
        __hip_atomic_store(&bar[0], 0u, __ATOMIC_RELAXED, __HIP_MEMORY_SCOPE_AGENT);
        __hip_atomic_fetch_add(&bar[1], 1u, __ATOMIC_RELEASE, __HIP_MEMORY_SCOPE_AGENT);
      } else {
        while (__hip_atomic_load(&bar[1], __ATOMIC_ACQUIRE, __HIP_MEMORY_SCOPE_AGENT) == gen) {
          __builtin_amdgcn_s_sleep(2);
        }
      }
    }
    __syncthreads();
  }
}

// ------------------------------------------------------------------- final FC
__global__ void fc_kernel(const unsigned short* __restrict__ seq,
                          const float* __restrict__ fcw,
                          const float* __restrict__ fcb,
                          float* __restrict__ out) {
  __shared__ float red[256];
  int b = blockIdx.x;
  const unsigned short* row = seq + ((size_t)b * T + (T - 1)) * H;
  float s = 0.f;
  for (int n = threadIdx.x; n < H; n += 256) s += bf2f(row[n]) * fcw[n];
  red[threadIdx.x] = s;
  __syncthreads();
  for (int off = 128; off > 0; off >>= 1) {
    if (threadIdx.x < off) red[threadIdx.x] += red[threadIdx.x + off];
    __syncthreads();
  }
  if (threadIdx.x == 0) out[b] = red[0] + fcb[0];
}

// ------------------------------------------------------------------ launcher
extern "C" void kernel_launch(void* const* d_in, const int* in_sizes, int n_in,
                              void* d_out, int out_size, void* d_ws, size_t ws_size,
                              hipStream_t stream) {
  const float* x    = (const float*)d_in[0];
  const float* Wih0 = (const float*)d_in[1];
  const float* Wih1 = (const float*)d_in[2];
  const float* Wih2 = (const float*)d_in[3];
  const float* Whh  = (const float*)d_in[4];
  const float* bih  = (const float*)d_in[5];
  const float* bhh  = (const float*)d_in[6];
  const float* fcw  = (const float*)d_in[7];
  const float* fcb  = (const float*)d_in[8];
  float* out = (float*)d_out;

  char* ws = (char*)d_ws;
  size_t off = 0;
  auto take = [&](size_t bytes) -> char* {
    char* p = ws + off;
    off = (off + bytes + 255) & ~(size_t)255;
    return p;
  };
  float*          xp   = (float*)take((size_t)B * T * G4H * 4);
  unsigned short* seqA = (unsigned short*)take((size_t)B * T * H * 2);
  unsigned short* seqB = (unsigned short*)take((size_t)B * T * H * 2);
  unsigned short* xbf  = (unsigned short*)take((size_t)B * T * IN * 2);
  unsigned short* w0   = (unsigned short*)take((size_t)G4H * IN * 2);
  unsigned short* w1   = (unsigned short*)take((size_t)G4H * H * 2);
  unsigned short* w2   = (unsigned short*)take((size_t)G4H * H * 2);
  unsigned short* whh  = (unsigned short*)take((size_t)3 * G4H * H * 2);
  unsigned short* hA   = (unsigned short*)take((size_t)B * H * 2);
  unsigned short* hB   = (unsigned short*)take((size_t)B * H * 2);
  float*          cbuf = (float*)take((size_t)B * H * 4);
  unsigned int*   bar  = (unsigned int*)take(256);

  auto cvt = [&](const float* s, unsigned short* d, int n) {
    cvt_f32_bf16<<<(n + 255) / 256, 256, 0, stream>>>(s, d, n);
  };
  cvt(x, xbf, B * T * IN);
  cvt(Wih0, w0, G4H * IN);
  cvt(Wih1, w1, G4H * H);
  cvt(Wih2, w2, G4H * H);
  cvt(Whh, whh, 3 * G4H * H);

  const unsigned short* wihs[3] = {w0, w1, w2};
  const unsigned short* ins[3]  = {xbf, seqA, seqB};
  unsigned short*       outs[3] = {seqA, seqB, seqA};
  const int Ks[3] = {IN, H, H};

  for (int l = 0; l < 3; ++l) {
    gemm_xp<<<dim3(G4H / 128, (B * T) / 64), 256, 0, stream>>>(
        ins[l], Ks[l], wihs[l], bih + l * G4H, bhh + l * G4H, xp);
    init_state<<<(B * H + 255) / 256, 256, 0, stream>>>(hA, hB, cbuf, bar);
    lstm_rec<<<NWG_REC, 256, 0, stream>>>(
        xp, whh + (size_t)l * G4H * H, hA, hB, cbuf, outs[l], bar);
  }
  fc_kernel<<<B, 256, 0, stream>>>(outs[2], fcw, fcb, out);

  (void)in_sizes; (void)n_in; (void)out_size; (void)ws_size;
}
